// MCT_86560770883935
// MI455X (gfx1250) — matrix operand order repaired and checked by reading, not compiled, not run on hardware
//
#include <hip/hip_runtime.h>
#include <math.h>

// ---------------------------------------------------------------------------
// Sizes from the reference
// ---------------------------------------------------------------------------
#define NN   1024
#define DD   128
#define RW_ITERS 30
static constexpr float INV2SIG2 = 0.5f / (15.0f * 15.0f);

typedef _Float16 v16h __attribute__((ext_vector_type(16)));
typedef _Float16 v8h  __attribute__((ext_vector_type(8)));
typedef float    v8f  __attribute__((ext_vector_type(8)));

// ---------------------------------------------------------------------------
// Kernel 1: compose the 3 affine expert layers into one: M = E2@E1@E0,
//           c = E2@(E1@b0 + b1) + b2.   One block, 128 threads.
// ---------------------------------------------------------------------------
__global__ __launch_bounds__(128) void mct_compose(const float* __restrict__ Ew,
                                                   const float* __restrict__ Eb,
                                                   float* __restrict__ M,
                                                   float* __restrict__ cvec) {
  __shared__ float T1[DD * DD];   // 64 KB (WGP has 320 KB LDS)
  __shared__ float tb[DD];
  const int t = threadIdx.x;
  const float* E0 = Ew;
  const float* E1 = Ew + DD * DD;
  const float* E2 = Ew + 2 * DD * DD;
  const float* B0 = Eb;
  const float* B1 = Eb + DD;
  const float* B2 = Eb + 2 * DD;

  // T1 = E1 @ E0  (thread t owns column t)
  for (int i = 0; i < DD; ++i) {
    float s = 0.f;
    for (int k = 0; k < DD; ++k) s = fmaf(E1[i * DD + k], E0[k * DD + t], s);
    T1[i * DD + t] = s;
  }
  { // tb = E1 @ B0 + B1
    float s = B1[t];
    for (int k = 0; k < DD; ++k) s = fmaf(E1[t * DD + k], B0[k], s);
    tb[t] = s;
  }
  __syncthreads();
  // M = E2 @ T1
  for (int i = 0; i < DD; ++i) {
    float s = 0.f;
    for (int k = 0; k < DD; ++k) s = fmaf(E2[i * DD + k], T1[k * DD + t], s);
    M[i * DD + t] = s;
  }
  { // c = E2 @ tb + B2
    float s = B2[t];
    for (int k = 0; k < DD; ++k) s = fmaf(E2[t * DD + k], tb[k], s);
    cvec[t] = s;
  }
}

// ---------------------------------------------------------------------------
// Kernel 2: per-row pipeline. x = f@W1.T + b1, u = M@x, then the per-column
// scalars of the quadratic: invmag = 1/||x||^2, a=||u||^2, b=2<u,c-x>,
// g=||c-x||^2. Also emits the f16 hi/lo split of x for the WMMA GEMM.
// 64 blocks x 128 threads, 16 rows per block. Deterministic tree reductions.
// ---------------------------------------------------------------------------
__global__ __launch_bounds__(128) void mct_rows(const float* __restrict__ f,
                                                const float* __restrict__ W1,
                                                const float* __restrict__ b1,
                                                const float* __restrict__ M,
                                                const float* __restrict__ cvec,
                                                _Float16* __restrict__ xh,
                                                _Float16* __restrict__ xl,
                                                float* __restrict__ invmag,
                                                float* __restrict__ aj,
                                                float* __restrict__ bj,
                                                float* __restrict__ gj) {
  __shared__ float sf[DD];
  __shared__ float sx[DD];
  __shared__ float red[4][DD];
  const int t = threadIdx.x;
  const float b1t = b1[t];
  const float ct  = cvec[t];
  const int n0 = blockIdx.x * 16;

  for (int n = n0; n < n0 + 16; ++n) {
    sf[t] = f[n * DD + t];
    __syncthreads();
    float xt = b1t;
    #pragma unroll 8
    for (int k = 0; k < DD; ++k) xt = fmaf(W1[t * DD + k], sf[k], xt);
    _Float16 xhi = (_Float16)xt;
    xh[n * DD + t] = xhi;
    xl[n * DD + t] = (_Float16)(xt - (float)xhi);
    sx[t] = xt;
    __syncthreads();
    float ut = 0.f;
    #pragma unroll 8
    for (int k = 0; k < DD; ++k) ut = fmaf(M[t * DD + k], sx[k], ut);
    float cm = ct - xt;
    red[0][t] = xt * xt;
    red[1][t] = ut * ut;
    red[2][t] = ut * cm;
    red[3][t] = cm * cm;
    __syncthreads();
    for (int s = 64; s > 0; s >>= 1) {
      if (t < s) {
        red[0][t] += red[0][t + s];
        red[1][t] += red[1][t + s];
        red[2][t] += red[2][t + s];
        red[3][t] += red[3][t + s];
      }
      __syncthreads();
    }
    if (t == 0) {
      invmag[n] = 1.0f / red[0][0];
      aj[n]     = red[1][0];
      bj[n]     = 2.0f * red[2][0];
      gj[n]     = red[3][0];
    }
    __syncthreads();
  }
}

// ---------------------------------------------------------------------------
// Kernel 3: WMMA scores GEMM fused with A = exp(-dist/2sig^2) and row sums.
// scores = x@x.T computed in split f16 precision: xh*xh + xh*xl + xl*xh,
// f32 accumulation via v_wmma_f32_16x16x32_f16.
// Grid: 64 blocks (16-row stripes), block = 128 threads = 4 waves.
// Wave w covers columns [w*256, w*256+256) as 16 tiles of 16x16.
//
// Fragment layouts used here (wave32; contiguous k, same order for A and B):
//  A 16x32 f16: lane L -> row m=L&15; half h -> K = kb + (L>>4)*16 + h
//               (two contiguous 16B loads)
//  B 32x16 f16: lane L -> col n=L&15; half h -> K = kb + (L>>4)*16 + h
//               (one contiguous 32B load)
//  C 16x16 f32: VGPR r, lane L -> row (L>>4)*8 + r, col L&15
// ---------------------------------------------------------------------------
__global__ __launch_bounds__(128) void mct_scores(const _Float16* __restrict__ xh,
                                                  const _Float16* __restrict__ xl,
                                                  const float* __restrict__ invmag,
                                                  const float* __restrict__ aj,
                                                  const float* __restrict__ bj,
                                                  const float* __restrict__ gj,
                                                  float* __restrict__ Amat,
                                                  float* __restrict__ dinv) {
  __shared__ float sums[16][64];
  const int lane = threadIdx.x & 31;
  const int wave = threadIdx.x >> 5;
  const int r0   = blockIdx.x * 16;   // stripe of 16 output rows
  const int m    = lane & 15;
  const int hi   = lane >> 4;

  // Load resident A fragments (stripe rows, K = 0..127 -> 4 k-steps).
  v16h ah[4], al[4];
  const int arow = r0 + m;
  #pragma unroll
  for (int kk = 0; kk < 4; ++kk) {
    const int kb = kk * 32 + hi * 16;
    union { v8h p[2]; v16h v; } uh, ul;
    uh.p[0] = *(const v8h*)(xh + arow * DD + kb);
    uh.p[1] = *(const v8h*)(xh + arow * DD + kb + 8);
    ul.p[0] = *(const v8h*)(xl + arow * DD + kb);
    ul.p[1] = *(const v8h*)(xl + arow * DD + kb + 8);
    ah[kk] = uh.v;
    al[kk] = ul.v;
  }

  float rowpart[8];
  #pragma unroll
  for (int r = 0; r < 8; ++r) rowpart[r] = 0.f;

  const int cbase = wave * 256;
  for (int tile = 0; tile < 16; ++tile) {
    const int jb = cbase + tile * 16;
    const int j  = jb + m;             // this lane's output column
    v8f c = {};
    #pragma unroll
    for (int kk = 0; kk < 4; ++kk) {
      const int kb = kk * 32 + hi * 16;
      v16h bhh = *(const v16h*)(xh + j * DD + kb);
      v16h bll = *(const v16h*)(xl + j * DD + kb);
      c = __builtin_amdgcn_wmma_f32_16x16x32_f16(false, ah[kk], false, bhh,
                                                 (short)0, c, false, false);
      c = __builtin_amdgcn_wmma_f32_16x16x32_f16(false, ah[kk], false, bll,
                                                 (short)0, c, false, false);
      c = __builtin_amdgcn_wmma_f32_16x16x32_f16(false, al[kk], false, bhh,
                                                 (short)0, c, false, false);
    }
    const float im = invmag[j];
    const float A_ = aj[j];
    const float B_ = bj[j];
    const float G_ = gj[j];
    #pragma unroll
    for (int r = 0; r < 8; ++r) {
      const int i = r0 + hi * 8 + r;
      const float S    = c[r] * im;
      float dist = fmaf(fmaf(A_, S, B_), S, G_);
      dist = (i == j) ? 0.f : dist;            // zeroed diagonal -> A_ii = 1
      const float Av = __expf(-dist * INV2SIG2);
      rowpart[r] += Av;
      Amat[i * NN + j] = Av;
    }
  }

  // Deterministic row-sum reduction: stage 64 partials per row, serial sum.
  #pragma unroll
  for (int r = 0; r < 8; ++r) sums[hi * 8 + r][wave * 16 + m] = rowpart[r];
  __syncthreads();
  if (threadIdx.x < 16) {
    float d = 0.f;
    for (int q = 0; q < 64; ++q) d += sums[threadIdx.x][q];
    dinv[r0 + threadIdx.x] = 1.0f / d;
  }
}

// ---------------------------------------------------------------------------
// Kernel 4: P0[j] = T[0][j] = A[0][j]/d[0] (0 on the diagonal).
// ---------------------------------------------------------------------------
__global__ void mct_initp(const float* __restrict__ Amat,
                          const float* __restrict__ dinv,
                          float* __restrict__ p0, float* __restrict__ pa) {
  const int j = blockIdx.x * blockDim.x + threadIdx.x;
  if (j < NN) {
    const float v = (j == 0) ? 0.f : Amat[j] * dinv[0];
    p0[j] = v;
    pa[j] = v;
  }
}

// ---------------------------------------------------------------------------
// Kernel 5 (x30): one random-walk step without materializing T:
//   (P@T)[j] = sum_i P[i]*A[i][j]/d[i] - P[j]/d[j]    (A_jj == 1)
//   Pnext    = 0.5*(P@T) + 0.5*P0
// Column-major matvec over A (4 MB, L2-resident). Coalesced row-wise reads.
// ---------------------------------------------------------------------------
__global__ __launch_bounds__(128) void mct_walk(const float* __restrict__ Amat,
                                                const float* __restrict__ dinv,
                                                const float* __restrict__ p0,
                                                const float* __restrict__ psrc,
                                                float* __restrict__ pdst) {
  const int j = blockIdx.x * 128 + threadIdx.x;
  float s = 0.f;
  #pragma unroll 8
  for (int i = 0; i < NN; ++i) s = fmaf(psrc[i] * dinv[i], Amat[i * NN + j], s);
  const float pj = psrc[j] * dinv[j];
  pdst[j] = 0.5f * (s - pj) + 0.5f * p0[j];
}

// ---------------------------------------------------------------------------
// Kernel 6: softmax over P[1:] -> out[0..1022]. One 1024-thread block.
// ---------------------------------------------------------------------------
__global__ __launch_bounds__(1024) void mct_softmax(const float* __restrict__ p,
                                                    float* __restrict__ out) {
  __shared__ float red[1024];
  const int t = threadIdx.x;
  const float v = (t >= 1) ? p[t] : -INFINITY;
  red[t] = v;
  __syncthreads();
  for (int s = 512; s > 0; s >>= 1) {
    if (t < s) red[t] = fmaxf(red[t], red[t + s]);
    __syncthreads();
  }
  const float mx = red[0];
  __syncthreads();
  const float e = (t >= 1) ? __expf(v - mx) : 0.f;
  red[t] = e;
  __syncthreads();
  for (int s = 512; s > 0; s >>= 1) {
    if (t < s) red[t] += red[t + s];
    __syncthreads();
  }
  const float inv = 1.0f / red[0];
  if (t >= 1) out[t - 1] = e * inv;
}

// ---------------------------------------------------------------------------
// Launcher
// ---------------------------------------------------------------------------
extern "C" void kernel_launch(void* const* d_in, const int* in_sizes, int n_in,
                              void* d_out, int out_size, void* d_ws, size_t ws_size,
                              hipStream_t stream) {
  const float* f  = (const float*)d_in[0];
  const float* W1 = (const float*)d_in[1];
  const float* b1 = (const float*)d_in[2];
  const float* Ew = (const float*)d_in[3];
  const float* Eb = (const float*)d_in[4];

  char* ws = (char*)d_ws;
  size_t off = 0;
  auto alloc = [&](size_t bytes) -> void* {
    void* p = ws + off;
    off += (bytes + 255) & ~size_t(255);
    return p;
  };

  float*    M      = (float*)alloc(DD * DD * sizeof(float));
  float*    cvec   = (float*)alloc(DD * sizeof(float));
  _Float16* xh     = (_Float16*)alloc((size_t)NN * DD * sizeof(_Float16));
  _Float16* xl     = (_Float16*)alloc((size_t)NN * DD * sizeof(_Float16));
  float*    invmag = (float*)alloc(NN * sizeof(float));
  float*    aj     = (float*)alloc(NN * sizeof(float));
  float*    bj     = (float*)alloc(NN * sizeof(float));
  float*    gj     = (float*)alloc(NN * sizeof(float));
  float*    Amat   = (float*)alloc((size_t)NN * NN * sizeof(float));
  float*    dinv   = (float*)alloc(NN * sizeof(float));
  float*    p0     = (float*)alloc(NN * sizeof(float));
  float*    pA     = (float*)alloc(NN * sizeof(float));
  float*    pB     = (float*)alloc(NN * sizeof(float));

  mct_compose<<<1, 128, 0, stream>>>(Ew, Eb, M, cvec);
  mct_rows<<<64, 128, 0, stream>>>(f, W1, b1, M, cvec, xh, xl, invmag, aj, bj, gj);
  mct_scores<<<64, 128, 0, stream>>>(xh, xl, invmag, aj, bj, gj, Amat, dinv);
  mct_initp<<<(NN + 255) / 256, 256, 0, stream>>>(Amat, dinv, p0, pA);
  for (int it = 0; it < RW_ITERS; ++it) {
    const float* src = (it & 1) ? pB : pA;
    float*       dst = (it & 1) ? pA : pB;
    mct_walk<<<NN / 128, 128, 0, stream>>>(Amat, dinv, p0, src, dst);
  }
  // RW_ITERS is even -> final P lives in pA
  mct_softmax<<<1, 1024, 0, stream>>>(pA, (float*)d_out);
}
